// SA_NET_61555471286984
// MI455X (gfx1250) — matrix operand order repaired and checked by reading, not compiled, not run on hardware
//
#include <hip/hip_runtime.h>
#include <hip/hip_bf16.h>

// ---------------------------------------------------------------------------
// MI455X (gfx1250, wave32) implementation.
//
//   1. LSTM input projection hoisted into one WMMA GEMM:
//        Xg[4096][2048] = inputs @ w_ih^T + b_ih + b_hh   (f16 in, f32 acc)
//   2. Sequential recurrence in a persistent cooperative kernel (32 WGs),
//      grid barrier per step, w_hh resident in L2, h staged through LDS.
//   3. CNN branch: convs as shifted WMMA GEMMs; pool+sigmoid fused f16 repack.
//   4. Tiny MLP head in one workgroup.
//
// GEMM kernel is templated on <NW, KW, Kc> so accumulators are static
// registers (no movrel indexing) and inner addressing is base+imm.
// ---------------------------------------------------------------------------

typedef __attribute__((ext_vector_type(16))) _Float16 v16h;
typedef __attribute__((ext_vector_type(8)))  _Float16 v8h;
typedef __attribute__((ext_vector_type(8)))  float    v8f;

#define H_SZ   512
#define T_SEQ  4096
#define E_IN   300
#define E_PAD  320    // K padded to multiple of 32 for wmma f16
#define G4H    2048   // 4*H
#define NWG_REC 32

__device__ __forceinline__ float sigf(float x) { return 1.f / (1.f + __expf(-x)); }

__device__ __forceinline__ v16h ld_frag16(const _Float16* p, int elem_gap) {
  union { v16h v; v8h h[2]; } u;
  u.h[0] = *(const v8h*)(p);
  u.h[1] = *(const v8h*)(p + elem_gap);
  return u.v;
}

// ---- Generic shifted-GEMM (plain GEMM when KW==1; 1-D convs otherwise) ----
// C[m][n] = bias(n) + sum_dt sum_k Apad[m+dt][k] * B[n][dt*Kc + k]
// Apad: f16 row-major, lda (rows include KW-1 halo rows).
// B: f16 [N][KW*Kc] row-major (== column-major WMMA B).
// Each wave computes a 16 x (16*NW) tile.
// Fragment layouts per ISA 7.12.2:
//   A 16x32: lane L -> row L&15, half-chunks of 8 at k0+(L>>4)*8 and +16.
//   B 32x16: lane L -> col  L&15, half-chunks of 8 at k0+(L>>4)*8 and +16.
template <int NW, int KW, int Kc>
__global__ void wmma_conv_gemm(const _Float16* __restrict__ Apad, int lda,
                               const _Float16* __restrict__ B,
                               float* __restrict__ C, int ldc,
                               int Mtiles, int Ntotal,
                               const float* __restrict__ bias,
                               const float* __restrict__ bias2) {
  int wave   = (blockIdx.x * blockDim.x + threadIdx.x) >> 5;
  int lane   = threadIdx.x & 31;
  int strips = Ntotal / (16 * NW);
  int mtile  = wave / strips;
  int nstrip = wave % strips;
  if (mtile >= Mtiles) return;            // wave-uniform; EXEC stays all-ones
  const int ldb = KW * Kc;

  int r  = lane & 15;
  int hi = lane >> 4;
  // per-lane base pointers hoisted out of the K loop
  const _Float16* aRow = Apad + (size_t)(mtile * 16 + r) * lda + hi * 8;
  const _Float16* bRow[NW];
#pragma unroll
  for (int j = 0; j < NW; ++j)
    bRow[j] = B + (size_t)(nstrip * (16 * NW) + j * 16 + r) * ldb + hi * 8;

  v8f acc[NW];
#pragma unroll
  for (int j = 0; j < NW; ++j) acc[j] = (v8f){0.f, 0.f, 0.f, 0.f, 0.f, 0.f, 0.f, 0.f};

#pragma unroll
  for (int dt = 0; dt < KW; ++dt) {
    const _Float16* aDt = aRow + dt * lda;
#pragma unroll
    for (int k0 = 0; k0 < Kc; k0 += 32) {
      v16h a = ld_frag16(aDt + k0, 16);
#pragma unroll
      for (int j = 0; j < NW; ++j) {
        v16h b = ld_frag16(bRow[j] + dt * Kc + k0, 16);
        acc[j] = __builtin_amdgcn_wmma_f32_16x16x32_f16(
            false, a, false, b, (short)0, acc[j], false, false);
      }
    }
  }

  // C/D layout: elem e of acc -> (m = mtile*16 + e + 8*hi, n = base + r)
  int mrow  = mtile * 16 + (hi << 3);
  int ncol0 = nstrip * (16 * NW) + r;
#pragma unroll
  for (int j = 0; j < NW; ++j) {
    int n = ncol0 + j * 16;
    float bs = bias ? bias[n] : 0.f;
    if (bias2) bs += bias2[n];
    float* cp = C + (size_t)mrow * ldc + n;
#pragma unroll
    for (int e = 0; e < 8; ++e) { *cp = acc[j][e] + bs; cp += ldc; }
  }
}

// ---- f32 -> f16 staging kernels ------------------------------------------
__global__ void cvt_pad_f16(const float* __restrict__ in, _Float16* __restrict__ out,
                            int rows, int kin, int kout) {
  int idx = blockIdx.x * blockDim.x + threadIdx.x;
  if (idx >= rows * kout) return;
  int r = idx / kout, k = idx - r * kout;
  out[idx] = (k < kin) ? (_Float16)in[(size_t)r * kin + k] : (_Float16)0.f;
}

// Aconv[518][320]: 3 zero halo rows, inputs[0..511] padded to 320, 3 zero rows.
__global__ void build_aconv(const float* __restrict__ in, _Float16* __restrict__ out) {
  int idx = blockIdx.x * blockDim.x + threadIdx.x;
  if (idx >= 518 * E_PAD) return;
  int r = idx / E_PAD, k = idx - r * E_PAD;
  _Float16 v = (_Float16)0.f;
  if (r >= 3 && r < 515 && k < E_IN) v = (_Float16)in[(size_t)(r - 3) * E_IN + k];
  out[idx] = v;
}

// [O][C][D] f32 -> [O][D][C] f16 (channel dim innermost for contiguous K)
__global__ void cvt_w_odc(const float* __restrict__ in, _Float16* __restrict__ out,
                          int O, int Cc, int D) {
  int idx = blockIdx.x * blockDim.x + threadIdx.x;
  if (idx >= O * Cc * D) return;
  int o = idx / (Cc * D);
  int rem = idx - o * (Cc * D);
  int d = rem / Cc, c = rem - d * Cc;
  out[idx] = (_Float16)in[(size_t)o * Cc * D + (size_t)c * D + d];
}

// ---- pool(2) + sigmoid, f32 [2*Lout][C] -> f16 [(Lout+2*pad)][C] w/ halo --
__global__ void pool_sig_f16(const float* __restrict__ in, int C, int Lout,
                             _Float16* __restrict__ out, int pad) {
  int idx = blockIdx.x * blockDim.x + threadIdx.x;
  int total = (Lout + 2 * pad) * C;
  if (idx >= total) return;
  int r = idx / C, c = idx - r * C;
  int l = r - pad;
  _Float16 v = (_Float16)0.f;
  if (l >= 0 && l < Lout) {
    float a = in[(size_t)(2 * l) * C + c];
    float b = in[(size_t)(2 * l + 1) * C + c];
    v = (_Float16)sigf(fmaxf(a, b));
  }
  out[idx] = v;
}

// conv4 out y4[64][16] -> sigmoid, flattened reference order yflat[o*64+l]
__global__ void sig_flatten(const float* __restrict__ y4, float* __restrict__ yflat) {
  int idx = blockIdx.x * blockDim.x + threadIdx.x;
  if (idx >= 1024) return;
  int o = idx >> 6, l = idx & 63;
  yflat[idx] = sigf(y4[(size_t)l * 16 + o]);
}

__global__ void zero_u32(unsigned* __restrict__ p, int n) {
  int i = blockIdx.x * blockDim.x + threadIdx.x;
  if (i < n) p[i] = 0u;
}

// ---- Persistent LSTM recurrence ------------------------------------------
// 32 WGs x 256 thr = 8192 thr = 512 hidden x 4 gates x 4 K-parts.
// hbuf: [2][512] double buffer in global; grid barrier per step.
__global__ void lstm_recurrence(const float* __restrict__ Xg,   // [T][2048]
                                const float* __restrict__ w_hh, // [2048][512]
                                float* __restrict__ hbuf,       // [2][512]
                                float* __restrict__ hsum,       // [512]
                                unsigned* __restrict__ cnt,
                                int T) {
  __shared__ float hsh[H_SZ];
  int gt   = blockIdx.x * blockDim.x + threadIdx.x;
  int j    = gt >> 4;        // hidden unit
  int sub  = gt & 15;
  int gate = sub >> 2;       // 0=i 1=f 2=g 3=o
  int part = sub & 3;        // K quarter
  int row  = gate * H_SZ + j;
  const float* wr = w_hh + (size_t)row * H_SZ + part * 128;

  float c = 0.f, hs = 0.f;
  for (int t = 0; t < T; ++t) {
    // stage h_t into LDS with device-coherent (volatile) loads
    for (int i = threadIdx.x; i < H_SZ; i += blockDim.x)
      hsh[i] = ((const volatile float*)hbuf)[(t & 1) * H_SZ + i];
    __syncthreads();

    if (t + 1 < T)  // hint next step's gate row toward L2 (global_prefetch)
      __builtin_prefetch(&Xg[(size_t)(t + 1) * G4H + row], 0, 1);

    float acc = 0.f;
    const float* hp = &hsh[part * 128];
#pragma unroll 8
    for (int q = 0; q < 128; q += 4) {
      float4 w4 = *(const float4*)(wr + q);
      acc = fmaf(w4.x, hp[q + 0], acc);
      acc = fmaf(w4.y, hp[q + 1], acc);
      acc = fmaf(w4.z, hp[q + 2], acc);
      acc = fmaf(w4.w, hp[q + 3], acc);
    }
    // reduce the 4 K-parts (consecutive lanes within the wave)
    acc += __shfl_xor(acc, 1, 32);
    acc += __shfl_xor(acc, 2, 32);
    float gpre = acc + Xg[(size_t)t * G4H + row];

    // gather i/f/g/o onto sub==0 lanes (subs 0,4,8,12 within each 16-group)
    float gi = gpre;
    float gf = __shfl_down(gpre, 4, 16);
    float gg = __shfl_down(gpre, 8, 16);
    float go = __shfl_down(gpre, 12, 16);
    if (sub == 0) {
      c = sigf(gf) * c + sigf(gi) * tanhf(gg);
      float hv = sigf(go) * tanhf(c);
      hs += hv;
      ((volatile float*)hbuf)[((t + 1) & 1) * H_SZ + j] = hv;
    }

    // grid barrier (monotonic counter; cnt zeroed at launch)
    __threadfence();
    __syncthreads();
    if (threadIdx.x == 0) {
      atomicAdd(cnt, 1u);
      unsigned target = (unsigned)(t + 1) * NWG_REC;
      while (atomicAdd(cnt, 0u) < target) __builtin_amdgcn_s_sleep(1);
    }
    __syncthreads();
  }
  if (sub == 0) hsum[j] = hs;
}

// ---- MLP head (single workgroup) -----------------------------------------
__global__ void mlp_head(const float* __restrict__ hsum, const float* __restrict__ yflat,
                         const float* __restrict__ fc1_w, const float* __restrict__ fc1_b,
                         const float* __restrict__ fc2_w, const float* __restrict__ fc2_b,
                         const float* __restrict__ fc3_w, const float* __restrict__ fc3_b,
                         float* __restrict__ out) {
  __shared__ float feat[1536];
  __shared__ float a1[128];
  __shared__ float a2[32];
  int tid = threadIdx.x;
  for (int i = tid; i < H_SZ; i += blockDim.x) feat[i] = hsum[i] * (1.f / 4096.f);
  for (int i = tid; i < 1024; i += blockDim.x) feat[H_SZ + i] = yflat[i];
  __syncthreads();
  if (tid < 128) {
    float s = fc1_b[tid];
    const float* w = fc1_w + (size_t)tid * 1536;
    for (int k = 0; k < 1536; ++k) s = fmaf(w[k], feat[k], s);
    a1[tid] = sigf(s);
  }
  __syncthreads();
  if (tid < 32) {
    float s = fc2_b[tid];
    const float* w = fc2_w + (size_t)tid * 128;
    for (int k = 0; k < 128; ++k) s = fmaf(w[k], a1[k], s);
    a2[tid] = sigf(s);
  }
  __syncthreads();
  if (tid == 0) {
    float s = fc3_b[0];
    for (int k = 0; k < 32; ++k) s = fmaf(fc3_w[k], a2[k], s);
    out[0] = sigf(s);
  }
}

// ---------------------------------------------------------------------------
static inline unsigned cdiv(unsigned a, unsigned b) { return (a + b - 1) / b; }

extern "C" void kernel_launch(void* const* d_in, const int* in_sizes, int n_in,
                              void* d_out, int out_size, void* d_ws, size_t ws_size,
                              hipStream_t stream) {
  (void)in_sizes; (void)n_in; (void)out_size; (void)ws_size;
  const float* inputs  = (const float*)d_in[0];
  const float* w_ih    = (const float*)d_in[1];
  const float* w_hh    = (const float*)d_in[2];
  const float* b_ih    = (const float*)d_in[3];
  const float* b_hh    = (const float*)d_in[4];
  const float* conv1_w = (const float*)d_in[5];
  const float* conv1_b = (const float*)d_in[6];
  const float* conv2_w = (const float*)d_in[7];
  const float* conv2_b = (const float*)d_in[8];
  const float* conv3_w = (const float*)d_in[9];
  const float* conv3_b = (const float*)d_in[10];
  const float* conv4_w = (const float*)d_in[11];
  const float* conv4_b = (const float*)d_in[12];
  const float* fc1_w   = (const float*)d_in[13];
  const float* fc1_b   = (const float*)d_in[14];
  const float* fc2_w   = (const float*)d_in[15];
  const float* fc2_b   = (const float*)d_in[16];
  const float* fc3_w   = (const float*)d_in[17];
  const float* fc3_b   = (const float*)d_in[18];
  float* out = (float*)d_out;

  // Workspace carve-up (256B aligned), ~40 MB total.
  char* ws = (char*)d_ws;
  size_t off = 0;
  auto take = [&](size_t bytes) -> char* {
    char* p = ws + off;
    off += (bytes + 255) & ~(size_t)255;
    return p;
  };
  _Float16* A16   = (_Float16*)take((size_t)T_SEQ * E_PAD * 2);   // inputs f16 padded
  _Float16* Wih16 = (_Float16*)take((size_t)G4H * E_PAD * 2);     // w_ih f16 padded
  float*    Xg    = (float*)   take((size_t)T_SEQ * G4H * 4);     // gate preacts
  _Float16* Aconv = (_Float16*)take((size_t)518 * E_PAD * 2);     // conv1 input + halo
  _Float16* W1h   = (_Float16*)take((size_t)256 * 7 * E_PAD * 2);
  float*    y1    = (float*)   take((size_t)512 * 256 * 4);
  _Float16* p1    = (_Float16*)take((size_t)260 * 256 * 2);       // pooled + halo(2)
  _Float16* W2h   = (_Float16*)take((size_t)64 * 5 * 256 * 2);
  float*    y2    = (float*)   take((size_t)256 * 64 * 4);
  _Float16* p2    = (_Float16*)take((size_t)130 * 64 * 2);        // pooled + halo(1)
  _Float16* W3h   = (_Float16*)take((size_t)256 * 3 * 64 * 2);
  float*    y3    = (float*)   take((size_t)128 * 256 * 4);
  _Float16* p3    = (_Float16*)take((size_t)64 * 256 * 2);        // pooled, no halo
  _Float16* W4h   = (_Float16*)take((size_t)16 * 256 * 2);
  float*    y4    = (float*)   take((size_t)64 * 16 * 4);
  float*    yflat = (float*)   take((size_t)1024 * 4);
  float*    hbuf  = (float*)   take((size_t)2 * H_SZ * 4);        // contiguous:
  float*    hsum  = (float*)   take((size_t)H_SZ * 4);            //   zeroed as one
  unsigned* cnt   = (unsigned*)take(256);                         //   region below

  const int TB = 256;
  // zero hbuf + hsum + cnt (contiguous: 1024 + 512 + 64 u32)
  zero_u32<<<cdiv(1600, TB), TB, 0, stream>>>((unsigned*)hbuf, 1600);

  // f16 staging
  cvt_pad_f16<<<cdiv(T_SEQ * E_PAD, TB), TB, 0, stream>>>(inputs, A16, T_SEQ, E_IN, E_PAD);
  build_aconv<<<cdiv(518 * E_PAD, TB), TB, 0, stream>>>(inputs, Aconv);
  cvt_pad_f16<<<cdiv(G4H * E_PAD, TB), TB, 0, stream>>>(w_ih, Wih16, G4H, E_IN, E_PAD);
  cvt_pad_f16<<<cdiv(256 * 7 * E_PAD, TB), TB, 0, stream>>>(conv1_w, W1h, 256 * 7, E_IN, E_PAD);
  cvt_w_odc<<<cdiv(64 * 256 * 5, TB), TB, 0, stream>>>(conv2_w, W2h, 64, 256, 5);
  cvt_w_odc<<<cdiv(256 * 64 * 3, TB), TB, 0, stream>>>(conv3_w, W3h, 256, 64, 3);
  cvt_pad_f16<<<cdiv(16 * 256, TB), TB, 0, stream>>>(conv4_w, W4h, 16, 256, 256);

  // Xg = inputs @ w_ih^T + b_ih + b_hh   (M=4096, N=2048, K=320)
  // waves = (4096/16) * (2048/64) = 8192 -> 1024 blocks of 8 waves
  wmma_conv_gemm<4, 1, E_PAD><<<1024, TB, 0, stream>>>(
      A16, E_PAD, Wih16, Xg, G4H, T_SEQ / 16, G4H, b_ih, b_hh);

  // CNN branch: conv1 (K=7x320) -> pool+sig -> conv2 (5x256) -> ... -> conv4
  wmma_conv_gemm<4, 7, E_PAD><<<16, TB, 0, stream>>>(
      Aconv, E_PAD, W1h, y1, 256, 32, 256, conv1_b, nullptr);
  pool_sig_f16<<<cdiv(260 * 256, TB), TB, 0, stream>>>(y1, 256, 256, p1, 2);
  wmma_conv_gemm<4, 5, 256><<<2, TB, 0, stream>>>(
      p1, 256, W2h, y2, 64, 16, 64, conv2_b, nullptr);
  pool_sig_f16<<<cdiv(130 * 64, TB), TB, 0, stream>>>(y2, 64, 128, p2, 1);
  wmma_conv_gemm<4, 3, 64><<<4, TB, 0, stream>>>(
      p2, 64, W3h, y3, 256, 8, 256, conv3_b, nullptr);
  pool_sig_f16<<<cdiv(64 * 256, TB), TB, 0, stream>>>(y3, 256, 64, p3, 0);
  wmma_conv_gemm<1, 1, 256><<<1, TB, 0, stream>>>(
      p3, 256, W4h, y4, 16, 4, 16, conv4_b, nullptr);
  sig_flatten<<<cdiv(1024, TB), TB, 0, stream>>>(y4, yflat);

  // Sequential LSTM recurrence (the latency-bound critical path)
  lstm_recurrence<<<NWG_REC, TB, 0, stream>>>(Xg, w_hh, hbuf, hsum, cnt, T_SEQ);

  // Head
  mlp_head<<<1, TB, 0, stream>>>(hsum, yflat, fc1_w, fc1_b, fc2_w, fc2_b,
                                 fc3_w, fc3_b, out);
}
